// _Int4Linear_54631984005366
// MI455X (gfx1250) — compile-verified, matrix operand order checked by reading, not run on hardware
//
#include <hip/hip_runtime.h>

typedef __attribute__((ext_vector_type(16))) _Float16 v16h;
typedef __attribute__((ext_vector_type(8)))  _Float16 v8h;
typedef __attribute__((ext_vector_type(2)))  _Float16 v2h;
typedef __attribute__((ext_vector_type(8)))  float    v8f;
typedef __attribute__((ext_vector_type(4)))  float    v4f;
typedef __attribute__((ext_vector_type(4)))  int      v4i;

#define K_DIM 4096
#define N_DIM 11008
#define M_DIM 4096
#define PACKW (K_DIM / 2)   // int32 words per output row (1 byte -> 2 nibbles each)

#define BM 128
#define BN 128
#define BK 32
#define LDT 40              // padded LDS row stride in halfs (80 B -> conflict-free b128 frag loads)
#define KSTEPS (K_DIM / BK) // 128 (even -> exact pair loop)

__global__ __launch_bounds__(256)
void _Int4Linear_wmma_kernel(const float* __restrict__ x,
                             const int*   __restrict__ wp,
                             const float* __restrict__ scale,
                             const float* __restrict__ bias,
                             float*       __restrict__ out)
{
    // double-buffered staging: one barrier per K-step
    __shared__ __align__(16) _Float16 lds_x[2][BM * LDT];
    __shared__ __align__(16) _Float16 lds_w[2][BN * LDT];

    const int tid  = threadIdx.x;
    const int wave = tid >> 5;
    const int lane = tid & 31;

    const int nblk = blockIdx.x * BN;
    const int mblk = blockIdx.y * BM;

    // 8 waves: 2 (M) x 4 (N); each wave computes 64x32 of the 128x128 block tile
    const int wm = (wave >> 2) * 64;
    const int wn = (wave & 3) * 32;

    // ---- global staging indexing: 2 threads per row, 16 halfs per thread ----
    const int lrow  = tid >> 1;   // 0..127
    const int lhalf = tid & 1;    // which 16-element chunk along K

    const float* xg = x  + (size_t)(mblk + lrow) * K_DIM + lhalf * 16;
    const int*   wg = wp + (size_t)(nblk + lrow) * PACKW + lhalf * 8;

    // ---- fragment addressing (16-bit A/B layout: lane = row, ksel picks K halves) ----
    const int frow = lane & 15;
    const int ksel = lane >> 4;

    v8f acc[4][2];
#pragma unroll
    for (int i = 0; i < 4; ++i)
#pragma unroll
        for (int j = 0; j < 2; ++j)
            acc[i][j] = (v8f){0.f, 0.f, 0.f, 0.f, 0.f, 0.f, 0.f, 0.f};

    const int st_x = lrow * LDT + lhalf * 16;   // LDS store offset (halfs)

    // --- stage one K-step (regs -> f16 -> LDS buffer `buf`) ---
    auto stage = [&](int buf, const v4f* xr, const v4i* wr) {
        // x: fp32 -> f16 (compiler packs with v_cvt_pk_f16_f32)
        union { _Float16 h[16]; v8h v[2]; } ux;
#pragma unroll
        for (int i = 0; i < 4; ++i)
#pragma unroll
            for (int j = 0; j < 4; ++j)
                ux.h[4 * i + j] = (_Float16)xr[i][j];
        v8h* dx = (v8h*)&lds_x[buf][st_x];
        dx[0] = ux.v[0];
        dx[1] = ux.v[1];

        // w: int4 nibbles -> f16 via exponent-bias trick:
        //   f16(0x6400 | n) == 1024 + n  (exact), so (pack | 0x64006400) - 1032 = {lo-8, hi-8}
        const v2h kbias = {(_Float16)1032.0f, (_Float16)1032.0f};
        union { v2h p[8]; v8h v[2]; } uw;
#pragma unroll
        for (int i = 0; i < 2; ++i)
#pragma unroll
            for (int j = 0; j < 4; ++j) {
                const unsigned u  = (unsigned)wr[i][j];
                const unsigned pk = (u & 0xFu) | ((u << 12) & 0xF0000u) | 0x64006400u;
                v2h h2;
                __builtin_memcpy(&h2, &pk, 4);
                uw.p[4 * i + j] = h2 - kbias;   // one v_pk_add_f16
            }
        v8h* dw = (v8h*)&lds_w[buf][st_x];
        dw[0] = uw.v[0];
        dw[1] = uw.v[1];
    };

    // --- one K-step of WMMAs from LDS buffer `buf` ---
    auto compute = [&](int buf) {
        v16h afrag[4], bfrag[2];
#pragma unroll
        for (int i = 0; i < 4; ++i) {
            const _Float16* p = &lds_x[buf][(wm + i * 16 + frow) * LDT + ksel * 8];
            v8h lo = *(const v8h*)p;
            v8h hi = *(const v8h*)(p + 16);
            afrag[i] = __builtin_shufflevector(lo, hi,
                0, 1, 2, 3, 4, 5, 6, 7, 8, 9, 10, 11, 12, 13, 14, 15);
        }
#pragma unroll
        for (int j = 0; j < 2; ++j) {
            const _Float16* p = &lds_w[buf][(wn + j * 16 + frow) * LDT + ksel * 8];
            v8h lo = *(const v8h*)p;
            v8h hi = *(const v8h*)(p + 16);
            bfrag[j] = __builtin_shufflevector(lo, hi,
                0, 1, 2, 3, 4, 5, 6, 7, 8, 9, 10, 11, 12, 13, 14, 15);
        }
#pragma unroll
        for (int i = 0; i < 4; ++i)
#pragma unroll
            for (int j = 0; j < 2; ++j)
                acc[i][j] = __builtin_amdgcn_wmma_f32_16x16x32_f16(
                    /*neg_a=*/false, afrag[i],
                    /*neg_b=*/false, bfrag[j],
                    /*c_mod=*/(short)0, acc[i][j],
                    /*reuse_a=*/false, /*reuse_b=*/false);
    };

    // two independent register sets -> no hand-off copies in the pair-unrolled loop
    v4f xa[4], xb[4];
    v4i wa[2], wb[2];

    // ---- pipeline prologue: buf0 <- G(0); xa/wa <- G(1) ----
    {
#pragma unroll
        for (int i = 0; i < 4; ++i) xa[i] = ((const v4f*)xg)[i];
#pragma unroll
        for (int i = 0; i < 2; ++i) wa[i] = ((const v4i*)wg)[i];
        stage(0, xa, wa);
#pragma unroll
        for (int i = 0; i < 4; ++i) xa[i] = ((const v4f*)(xg + BK))[i];
#pragma unroll
        for (int i = 0; i < 2; ++i) wa[i] = ((const v4i*)(wg + BK / 2))[i];
        __syncthreads();
    }

    // prefetch pointer, pointing at step kt+2 data; one K-step stride per advance
    const float* xg_n = xg + 2 * BK;
    const int*   wg_n = wg + BK;

    // Invariant at loop top (kt even): buf0 holds step kt, xa/wa hold step kt+1.
    for (int kt = 0; kt < KSTEPS; kt += 2) {
        // ================= half A: compute step kt from buf0 =================
        if (kt + 2 < KSTEPS) {   // prefetch G(kt+2) into set B
#pragma unroll
            for (int i = 0; i < 4; ++i) xb[i] = ((const v4f*)xg_n)[i];
#pragma unroll
            for (int i = 0; i < 2; ++i) wb[i] = ((const v4i*)wg_n)[i];
            xg_n += BK;
            wg_n += BK / 2;
        }
        stage(1, xa, wa);        // kt+1 -> buf1 (always valid: kt <= KSTEPS-2)
        compute(0);
        __syncthreads();

        // ================ half B: compute step kt+1 from buf1 ================
        if (kt + 3 < KSTEPS) {   // prefetch G(kt+3) into set A
#pragma unroll
            for (int i = 0; i < 4; ++i) xa[i] = ((const v4f*)xg_n)[i];
#pragma unroll
            for (int i = 0; i < 2; ++i) wa[i] = ((const v4i*)wg_n)[i];
            xg_n += BK;
            wg_n += BK / 2;
        }
        if (kt + 2 < KSTEPS) stage(0, xb, wb);   // kt+2 -> buf0
        compute(1);
        __syncthreads();
    }

    // ---- epilogue: out = acc * scale[n] + bias[n] ----
    // C/D layout: VGPR r, lanes 0-15 -> M=r, N=lane; lanes 16-31 -> M=8+r, N=lane-16
#pragma unroll
    for (int j = 0; j < 2; ++j) {
        const int n = nblk + wn + j * 16 + frow;
        const float s = scale[n];
        const float b = bias[n];
#pragma unroll
        for (int i = 0; i < 4; ++i) {
            const int mbase = mblk + wm + i * 16 + ksel * 8;
#pragma unroll
            for (int r = 0; r < 8; ++r)
                out[(size_t)(mbase + r) * N_DIM + n] = acc[i][j][r] * s + b;
        }
    }
}

extern "C" void kernel_launch(void* const* d_in, const int* in_sizes, int n_in,
                              void* d_out, int out_size, void* d_ws, size_t ws_size,
                              hipStream_t stream) {
    const float* x     = (const float*)d_in[0];
    const int*   wp    = (const int*)d_in[1];
    const float* scale = (const float*)d_in[2];
    const float* bias  = (const float*)d_in[3];
    float*       out   = (float*)d_out;

    dim3 grid(N_DIM / BN, M_DIM / BM);  // (86, 32)
    dim3 block(256);
    _Int4Linear_wmma_kernel<<<grid, block, 0, stream>>>(x, wp, scale, bias, out);
}
